// SDTA_68238440399246
// MI455X (gfx1250) — matrix operand order repaired and checked by reading, not compiled, not run on hardware
//
#include <hip/hip_runtime.h>
#include <hip/hip_bf16.h>

typedef _Float16 half_t;
typedef __attribute__((ext_vector_type(16))) _Float16 v16h;
typedef __attribute__((ext_vector_type(8)))  _Float16 v8h;
typedef __attribute__((ext_vector_type(8)))  float    v8f;

// ---------------- constants -------------------------------------------------
#define BATCH   16
#define DIM     384
#define NH      8
#define HD      48
#define SPLIT   128
#define HW      56
#define NTOK    3136            // 56*56
#define MTOT    (BATCH*NTOK)    // 50176
#define QKVN    1152
#define R1      192
#define HID     1536

// ---------------- workspace layout (bytes, all 256-aligned) -----------------
#define OFF_WQKV  ((size_t)0)           // f16 [1152,384]
#define OFF_WPROJ ((size_t)884736)      // f16 [384,384]
#define OFF_WM1D  ((size_t)1179648)     // f16 [192,384]
#define OFF_WM1U  ((size_t)1327104)     // f16 [1536,192]
#define OFF_WM2D  ((size_t)1916928)     // f16 [192,1536]
#define OFF_WM2U  ((size_t)2506752)     // f16 [384,192]
#define OFF_R4    ((size_t)2654208)     // s1p/s2p f32 (phase1) ; qn/kn f16 (phase2)
#define OFF_S1P   (OFF_R4)
#define OFF_S2P   (OFF_R4 + (size_t)25690112)
#define OFF_QN    (OFF_R4)
#define OFF_KN    (OFF_R4 + (size_t)38535168)
#define OFF_XF    ((size_t)79724544)    // f32 [50176,384]  x_flat -> xres (in place)
#define OFF_Y     ((size_t)156794880)   // f16 [50176,384]  yln -> yattn
#define OFF_BIG   ((size_t)195330048)   // f16 qkv[50176,1152] -> z[50176,384] -> t2[50176,1536]
#define OFF_LG    ((size_t)349470720)   // f32 [128,48,48]
#define OFF_AT    ((size_t)350650368)   // f16 [128,48,64]
#define OFF_T1    ((size_t)351436800)   // f16 [50176,192]  t1 -> t3

// ---------------- WMMA helpers ---------------------------------------------
union V16H { v16h v; v8h h[2]; };

__device__ __forceinline__ v16h load_frag16(const half_t* p0, const half_t* p1) {
  V16H u;
  u.h[0] = *(const v8h*)p0;
  u.h[1] = *(const v8h*)p1;
  return u.v;
}

__device__ __forceinline__ v8f wmma_f16(v16h a, v16h b, v8f c) {
  return __builtin_amdgcn_wmma_f32_16x16x32_f16(false, a, false, b, (short)0, c,
                                                false, false);
}

__device__ __forceinline__ float gelu_exact(float x) {
  return 0.5f * x * (1.0f + erff(x * 0.70710678118654752f));
}

// ---------------- weight prep: f32 [K,N] -> f16 [N,K] -----------------------
__global__ __launch_bounds__(256)
void prep_w(const float* __restrict__ in, half_t* __restrict__ out, int K, int N) {
  int idx = blockIdx.x * 256 + threadIdx.x;
  if (idx < K * N) {
    int k = idx / N, n = idx - k * N;
    out[(size_t)n * K + k] = (half_t)in[idx];
  }
}

// ---------------- depthwise 3x3 (SAME), in = A + B --------------------------
__global__ __launch_bounds__(256)
void dwconv3x3(const float* __restrict__ pA, long bStrideA,
               const float* __restrict__ pB, long bStrideB,
               const float* __restrict__ w, const float* __restrict__ bias,
               float* __restrict__ out) {
  int bc = blockIdx.x;             // b*128 + c
  int b = bc >> 7, c = bc & 127;
  const float* a  = pA + (size_t)b * bStrideA + (size_t)c * NTOK;
  const float* bb = pB + (size_t)b * bStrideB + (size_t)c * NTOK;
  float* o = out + (size_t)bc * NTOK;
  float wg[9];
#pragma unroll
  for (int i = 0; i < 9; i++) wg[i] = w[c * 9 + i];
  float bv = bias[c];
  for (int p = threadIdx.x; p < NTOK; p += 256) {
    int y = p / HW, x = p - y * HW;
    float acc = bv;
#pragma unroll
    for (int dy = -1; dy <= 1; dy++) {
      int yy = y + dy;
      if (yy < 0 || yy >= HW) continue;
#pragma unroll
      for (int dx = -1; dx <= 1; dx++) {
        int xx = x + dx;
        if (xx < 0 || xx >= HW) continue;
        int q = yy * HW + xx;
        acc += wg[(dy + 1) * 3 + (dx + 1)] * (a[q] + bb[q]);
      }
    }
    o[p] = acc;
  }
}

// ---------------- pack + LN1: NCHW pieces -> x_flat f32, yln f16 ------------
__global__ __launch_bounds__(128)
void pack_ln1(const float* __restrict__ x, const float* __restrict__ s1p,
              const float* __restrict__ s2p, const float* __restrict__ g,
              const float* __restrict__ be, float* __restrict__ xflat,
              half_t* __restrict__ yln) {
  __shared__ float red[128];
  int m = blockIdx.x;
  int b = m / NTOK, n = m - b * NTOK;
  int t = threadIdx.x;
  float vals[3];
#pragma unroll
  for (int j = 0; j < 3; j++) {
    int c = t + j * 128;
    float v;
    if (c < 128)      v = x  [((size_t)b * DIM + c) * NTOK + n];
    else if (c < 256) v = s1p[((size_t)b * SPLIT + (c - 128)) * NTOK + n];
    else              v = s2p[((size_t)b * SPLIT + (c - 256)) * NTOK + n];
    vals[j] = v;
  }
  float s = vals[0] + vals[1] + vals[2];
  red[t] = s; __syncthreads();
  for (int o = 64; o > 0; o >>= 1) { if (t < o) red[t] += red[t + o]; __syncthreads(); }
  float mean = red[0] * (1.0f / DIM); __syncthreads();
  float q = 0.f;
#pragma unroll
  for (int j = 0; j < 3; j++) { float d = vals[j] - mean; q += d * d; }
  red[t] = q; __syncthreads();
  for (int o = 64; o > 0; o >>= 1) { if (t < o) red[t] += red[t + o]; __syncthreads(); }
  float rstd = rsqrtf(red[0] * (1.0f / DIM) + 1e-6f);
#pragma unroll
  for (int j = 0; j < 3; j++) {
    int c = t + j * 128;
    float xv = vals[j];
    xflat[(size_t)m * DIM + c] = xv;
    yln[(size_t)m * DIM + c] = (half_t)((xv - mean) * rstd * g[c] + be[c]);
  }
}

// ---------------- LN over contiguous f32 rows -> f16 ------------------------
__global__ __launch_bounds__(128)
void ln_rows(const float* __restrict__ in, const float* __restrict__ g,
             const float* __restrict__ be, half_t* __restrict__ out) {
  __shared__ float red[128];
  int m = blockIdx.x, t = threadIdx.x;
  float vals[3];
#pragma unroll
  for (int j = 0; j < 3; j++) vals[j] = in[(size_t)m * DIM + t + j * 128];
  float s = vals[0] + vals[1] + vals[2];
  red[t] = s; __syncthreads();
  for (int o = 64; o > 0; o >>= 1) { if (t < o) red[t] += red[t + o]; __syncthreads(); }
  float mean = red[0] * (1.0f / DIM); __syncthreads();
  float q = 0.f;
#pragma unroll
  for (int j = 0; j < 3; j++) { float d = vals[j] - mean; q += d * d; }
  red[t] = q; __syncthreads();
  for (int o = 64; o > 0; o >>= 1) { if (t < o) red[t] += red[t + o]; __syncthreads(); }
  float rstd = rsqrtf(red[0] * (1.0f / DIM) + 1e-6f);
#pragma unroll
  for (int j = 0; j < 3; j++) {
    int c = t + j * 128;
    out[(size_t)m * DIM + c] = (half_t)((vals[j] - mean) * rstd * g[c] + be[c]);
  }
}

// ---------------- generic WMMA GEMM, 32x64 tile per wave --------------------
// A: f16 [M,K] row-major (lda=K). BT: f16 [N,K] (pre-transposed weights).
// Each B fragment feeds two WMMAs (two 16-row groups): 8 wmma / 12 b128 loads.
// MODE 0: store f16 C            MODE 1: bias+GELU -> f16
// MODE 2: bias+resid -> f32 (in place ok)
// MODE 3: bias+resid -> f32 NCHW-transposed (final output)
template <int MODE>
__global__ __launch_bounds__(256)
void gemm16(const half_t* __restrict__ A, const half_t* __restrict__ BT,
            half_t* __restrict__ Oh, float* __restrict__ Of,
            const float* __restrict__ bias, const float* __restrict__ resid,
            int M, int N, int K, int totalWaves) {
  int wave = blockIdx.x * 8 + (threadIdx.x >> 5);
  if (wave >= totalWaves) return;
  int lane = threadIdx.x & 31;
  int r = lane & 15, hi = lane >> 4;
  int nt4 = N >> 6;
  int mt = wave / nt4;                 // 32-row tiles
  int n0 = (wave - mt * nt4) << 6;
  const half_t* arow0 = A + (size_t)(mt * 32 + r) * K;
  const half_t* arow1 = arow0 + (size_t)16 * K;
  const half_t* bcol = BT + (size_t)(n0 + r) * K;
  v8f acc[2][4] = {};
  for (int k0 = 0; k0 < K; k0 += 32) {
    __builtin_prefetch(arow0 + k0 + 128, 0, 1);
    __builtin_prefetch(arow1 + k0 + 128, 0, 1);
    v16h a0 = load_frag16(arow0 + k0 + hi * 8, arow0 + k0 + hi * 8 + 16);
    v16h a1 = load_frag16(arow1 + k0 + hi * 8, arow1 + k0 + hi * 8 + 16);
#pragma unroll
    for (int j = 0; j < 4; j++) {
      const half_t* bp = bcol + (size_t)(j * 16) * K + k0 + hi * 8;
      v16h bf = load_frag16(bp, bp + 16);
      acc[0][j] = wmma_f16(a0, bf, acc[0][j]);
      acc[1][j] = wmma_f16(a1, bf, acc[1][j]);
    }
  }
#pragma unroll
  for (int g = 0; g < 2; g++) {
    int tokBase = mt * 32 + g * 16 + hi * 8;
#pragma unroll
    for (int j = 0; j < 4; j++) {
      int col = n0 + j * 16 + r;
      float bb = (MODE != 0) ? bias[col] : 0.0f;
#pragma unroll
      for (int v = 0; v < 8; v++) {
        int tok = tokBase + v;
        float val = acc[g][j][v];
        if (MODE == 0) {
          Oh[(size_t)tok * N + col] = (half_t)val;
        } else if (MODE == 1) {
          Oh[(size_t)tok * N + col] = (half_t)gelu_exact(val + bb);
        } else if (MODE == 2) {
          size_t idx = (size_t)tok * N + col;
          Of[idx] = resid[idx] + val + bb;
        } else {  // MODE 3: out[b, col, n] NCHW
          int b = tok / NTOK, n = tok - b * NTOK;
          float rres = resid[(size_t)tok * N + col];
          Of[((size_t)b * DIM + col) * NTOK + n] = rres + val + bb;
        }
      }
    }
  }
}

// ---------------- fold L2-norm + transpose q,k -> [bh, HD, NTOK] f16 --------
__global__ __launch_bounds__(256)
void norm_transpose(const half_t* __restrict__ qkv, half_t* __restrict__ qn,
                    half_t* __restrict__ kn) {
  __shared__ float red[256];
  int id = blockIdx.x;             // 12288 = 2 * 128 * 48
  int which = id & 1;
  int rest = id >> 1;
  int bh = rest / HD, d = rest - bh * HD;
  int b = bh >> 3, h = bh & 7;
  int col = which * DIM + h * HD + d;
  const half_t* src = qkv + (size_t)b * NTOK * QKVN + col;
  int t = threadIdx.x;
  float s = 0.f;
  for (int n = t; n < NTOK; n += 256) {
    float v = (float)src[(size_t)n * QKVN];
    s += v * v;
  }
  red[t] = s; __syncthreads();
  for (int o = 128; o > 0; o >>= 1) { if (t < o) red[t] += red[t + o]; __syncthreads(); }
  float scale = 1.0f / fmaxf(sqrtf(red[0]), 1e-12f);
  half_t* dst = (which ? kn : qn) + (size_t)bh * HD * NTOK + (size_t)d * NTOK;
  for (int n = t; n < NTOK; n += 256)
    dst[n] = (half_t)((float)src[(size_t)n * QKVN] * scale);
}

// ---------------- attention logits: [48,48] = qn @ kn^T, K=3136 -------------
__global__ __launch_bounds__(288)
void attn_qk(const half_t* __restrict__ qn, const half_t* __restrict__ kn,
             float* __restrict__ logits) {
  int bh = blockIdx.x;
  int wave = threadIdx.x >> 5;
  int lane = threadIdx.x & 31;
  int r = lane & 15, hi = lane >> 4;
  int mt = wave / 3, nt = wave - mt * 3;
  const half_t* arow = qn + (size_t)bh * HD * NTOK + (size_t)(mt * 16 + r) * NTOK;
  const half_t* brow = kn + (size_t)bh * HD * NTOK + (size_t)(nt * 16 + r) * NTOK;
  v8f acc = {};
  for (int k0 = 0; k0 < NTOK; k0 += 32) {
    __builtin_prefetch(arow + k0 + 128, 0, 1);
    v16h a = load_frag16(arow + k0 + hi * 8, arow + k0 + hi * 8 + 16);
    v16h b = load_frag16(brow + k0 + hi * 8, brow + k0 + hi * 8 + 16);
    acc = wmma_f16(a, b, acc);
  }
#pragma unroll
  for (int v = 0; v < 8; v++)
    logits[(size_t)bh * (HD * HD) + (size_t)(mt * 16 + hi * 8 + v) * HD + nt * 16 + r] = acc[v];
}

// ---------------- softmax rows of 48, pad to K=64 with zeros ----------------
__global__ __launch_bounds__(64)
void softmax48(const float* __restrict__ logits, half_t* __restrict__ attn) {
  int bh = blockIdx.x, t = threadIdx.x;
  if (t >= HD) return;
  const float* row = logits + (size_t)bh * (HD * HD) + (size_t)t * HD;
  float mx = -1e30f;
#pragma unroll
  for (int i = 0; i < HD; i++) mx = fmaxf(mx, row[i]);
  float e[HD]; float s = 0.f;
#pragma unroll
  for (int i = 0; i < HD; i++) { e[i] = __expf(row[i] - mx); s += e[i]; }
  float inv = 1.0f / s;
  half_t* dst = attn + (size_t)bh * (HD * 64) + (size_t)t * 64;
#pragma unroll
  for (int i = 0; i < HD; i++) dst[i] = (half_t)(e[i] * inv);
#pragma unroll
  for (int i = HD; i < 64; i++) dst[i] = (half_t)0.0f;
}

// ---------------- y = attn @ v (v read in-place from qkv f16) ---------------
__global__ __launch_bounds__(256)
void attn_v(const half_t* __restrict__ attn, const half_t* __restrict__ qkv,
            half_t* __restrict__ yattn, int totalWaves) {
  int wave = blockIdx.x * 8 + (threadIdx.x >> 5);
  if (wave >= totalWaves) return;       // 128 * 588
  int lane = threadIdx.x & 31;
  int r = lane & 15, hi = lane >> 4;
  int bh = wave / 588;
  int tt = wave - bh * 588;
  int mt = tt / 196, nt = tt - mt * 196;
  int b = bh >> 3, h = bh & 7;
  const half_t* arow = attn + (size_t)bh * (HD * 64) + (size_t)(mt * 16 + r) * 64;
  const half_t* brow = qkv + (size_t)b * NTOK * QKVN + 2 * DIM + h * HD +
                       (size_t)(nt * 16 + r) * QKVN;
  v8f acc = {};
#pragma unroll
  for (int k0 = 0; k0 < 64; k0 += 32) {
    v16h a = load_frag16(arow + k0 + hi * 8, arow + k0 + hi * 8 + 16);
    v16h bf = load_frag16(brow + k0 + hi * 8, brow + k0 + hi * 8 + 16);
    acc = wmma_f16(a, bf, acc);
  }
  // channel dim (rows of this gemm) is contiguous in yattn -> one v8h store
  v8h hv;
#pragma unroll
  for (int v = 0; v < 8; v++) hv[v] = (half_t)acc[v];
  half_t* dst = yattn + (size_t)(b * NTOK + nt * 16 + r) * DIM + h * HD + mt * 16 + hi * 8;
  *(v8h*)dst = hv;
}

// ---------------- host-side launch ------------------------------------------
extern "C" void kernel_launch(void* const* d_in, const int* in_sizes, int n_in,
                              void* d_out, int out_size, void* d_ws, size_t ws_size,
                              hipStream_t stream) {
  const float* x       = (const float*)d_in[0];
  const float* dw_w0   = (const float*)d_in[1];
  const float* dw_b0   = (const float*)d_in[2];
  const float* dw_w1   = (const float*)d_in[3];
  const float* dw_b1   = (const float*)d_in[4];
  const float* ln1_g   = (const float*)d_in[5];
  const float* ln1_b   = (const float*)d_in[6];
  const float* qkv_w   = (const float*)d_in[7];
  const float* proj_w  = (const float*)d_in[8];
  const float* proj_b  = (const float*)d_in[9];
  const float* ln2_g   = (const float*)d_in[10];
  const float* ln2_b   = (const float*)d_in[11];
  const float* mlp1_dw = (const float*)d_in[12];
  const float* mlp1_uw = (const float*)d_in[13];
  const float* mlp1_ub = (const float*)d_in[14];
  const float* mlp2_dw = (const float*)d_in[15];
  const float* mlp2_uw = (const float*)d_in[16];
  const float* mlp2_ub = (const float*)d_in[17];

  char* ws = (char*)d_ws;
  half_t* wqkvT  = (half_t*)(ws + OFF_WQKV);
  half_t* wprojT = (half_t*)(ws + OFF_WPROJ);
  half_t* wm1dT  = (half_t*)(ws + OFF_WM1D);
  half_t* wm1uT  = (half_t*)(ws + OFF_WM1U);
  half_t* wm2dT  = (half_t*)(ws + OFF_WM2D);
  half_t* wm2uT  = (half_t*)(ws + OFF_WM2U);
  float*  s1p    = (float*)(ws + OFF_S1P);
  float*  s2p    = (float*)(ws + OFF_S2P);
  half_t* qn     = (half_t*)(ws + OFF_QN);
  half_t* kn     = (half_t*)(ws + OFF_KN);
  float*  xflat  = (float*)(ws + OFF_XF);     // x_flat, then xres (in place)
  half_t* ybuf   = (half_t*)(ws + OFF_Y);     // yln, then yattn
  half_t* qkvh   = (half_t*)(ws + OFF_BIG);   // qkv, then z, then t2
  float*  logits = (float*)(ws + OFF_LG);
  half_t* attnh  = (half_t*)(ws + OFF_AT);
  half_t* t1     = (half_t*)(ws + OFF_T1);    // t1, then t3
  float*  out    = (float*)d_out;

  // 0) weights -> f16, transposed to [N,K]
  prep_w<<<(DIM * QKVN + 255) / 256, 256, 0, stream>>>(qkv_w,  wqkvT,  DIM, QKVN);
  prep_w<<<(DIM * DIM  + 255) / 256, 256, 0, stream>>>(proj_w, wprojT, DIM, DIM);
  prep_w<<<(DIM * R1   + 255) / 256, 256, 0, stream>>>(mlp1_dw, wm1dT, DIM, R1);
  prep_w<<<(R1 * HID   + 255) / 256, 256, 0, stream>>>(mlp1_uw, wm1uT, R1, HID);
  prep_w<<<(HID * R1   + 255) / 256, 256, 0, stream>>>(mlp2_dw, wm2dT, HID, R1);
  prep_w<<<(R1 * DIM   + 255) / 256, 256, 0, stream>>>(mlp2_uw, wm2uT, R1, DIM);

  // 1-2) depthwise chain: s1' = dw(s0+s1); s2' = dw(s2+s1')
  dwconv3x3<<<BATCH * SPLIT, 256, 0, stream>>>(
      x, (long)DIM * NTOK, x + (size_t)SPLIT * NTOK, (long)DIM * NTOK,
      dw_w0, dw_b0, s1p);
  dwconv3x3<<<BATCH * SPLIT, 256, 0, stream>>>(
      x + (size_t)2 * SPLIT * NTOK, (long)DIM * NTOK, s1p, (long)SPLIT * NTOK,
      dw_w1, dw_b1, s2p);

  // 3) pack NCHW -> token-major + LN1
  pack_ln1<<<MTOT, 128, 0, stream>>>(x, s1p, s2p, ln1_g, ln1_b, xflat, ybuf);

  // 4) qkv GEMM  [50176,384] x [384,1152] -> f16
  {
    int waves = (MTOT / 32) * (QKVN / 64);
    gemm16<0><<<(waves + 7) / 8, 256, 0, stream>>>(
        ybuf, wqkvT, qkvh, nullptr, nullptr, nullptr, MTOT, QKVN, DIM, waves);
  }

  // 5) L2-normalize + transpose q,k  (overwrites s1p/s2p region — now dead)
  norm_transpose<<<2 * BATCH * NH * HD, 256, 0, stream>>>(qkvh, qn, kn);

  // 6) attention logits 48x48, K=3136
  attn_qk<<<BATCH * NH, 288, 0, stream>>>(qn, kn, logits);

  // 7) softmax, pad A to K=64
  softmax48<<<BATCH * NH, 64, 0, stream>>>(logits, attnh);

  // 8) y = attn @ v  (v read in place from qkv f16)
  {
    int waves = BATCH * NH * (HD / 16) * (NTOK / 16);   // 128*3*196 = 75264
    attn_v<<<(waves + 7) / 8, 256, 0, stream>>>(attnh, qkvh, ybuf, waves);
  }

  // 9) proj GEMM + bias + residual (in place into xflat -> xres)
  {
    int waves = (MTOT / 32) * (DIM / 64);
    gemm16<2><<<(waves + 7) / 8, 256, 0, stream>>>(
        ybuf, wprojT, nullptr, xflat, proj_b, xflat, MTOT, DIM, DIM, waves);
  }

  // 10) LN2 -> z f16 (reuses qkv region)
  ln_rows<<<MTOT, 128, 0, stream>>>(xflat, ln2_g, ln2_b, qkvh);

  // 11) t1 = z @ mlp1_dw   [50176,384]x[384,192]
  {
    int waves = (MTOT / 32) * (R1 / 64);
    gemm16<0><<<(waves + 7) / 8, 256, 0, stream>>>(
        qkvh, wm1dT, t1, nullptr, nullptr, nullptr, MTOT, R1, DIM, waves);
  }

  // 12) t2 = gelu(t1 @ mlp1_uw + b)  [50176,192]x[192,1536] (t2 reuses OFF_BIG)
  {
    int waves = (MTOT / 32) * (HID / 64);
    gemm16<1><<<(waves + 7) / 8, 256, 0, stream>>>(
        t1, wm1uT, qkvh, nullptr, mlp1_ub, nullptr, MTOT, HID, R1, waves);
  }

  // 13) t3 = t2 @ mlp2_dw  [50176,1536]x[1536,192]  (t3 reuses t1)
  {
    int waves = (MTOT / 32) * (R1 / 64);
    gemm16<0><<<(waves + 7) / 8, 256, 0, stream>>>(
        qkvh, wm2dT, t1, nullptr, nullptr, nullptr, MTOT, R1, HID, waves);
  }

  // 14) out = xres + t3 @ mlp2_uw + b, stored NCHW fp32
  {
    int waves = (MTOT / 32) * (DIM / 64);
    gemm16<3><<<(waves + 7) / 8, 256, 0, stream>>>(
        t1, wm2uT, nullptr, out, mlp2_ub, xflat, MTOT, DIM, R1, waves);
  }
}
